// MultiHeadAttention_43026982371491
// MI455X (gfx1250) — matrix operand order repaired and checked by reading, not compiled, not run on hardware
//
#include <hip/hip_runtime.h>
#include <hip/hip_bf16.h>

typedef __attribute__((ext_vector_type(16))) _Float16 v16h;
typedef __attribute__((ext_vector_type(8)))  float    v8f;

#define NHEADS 16
#define HDIM   128
#define DIMSZ  2048
#define SEQ    2048
#define BATCH  2

// ---- WMMA operand loaders (ISA 7.12.2 layouts, wave32) -------------------
// A matrix 16x32 f16: lane L (m=L&15, g=L>>4) holds halves:
//   h[0..7]  = A(m, k0 + 8g + j)
//   h[8..15] = A(m, k0 + 16 + 8g + j)
__device__ __forceinline__ v16h load_a16(const _Float16* __restrict__ base,
                                         int ld, int m, int g, int k0) {
    const _Float16* p = base + (size_t)m * ld + k0 + 8 * g;
    v16h r;
#pragma unroll
    for (int j = 0; j < 8; ++j) r[j] = p[j];
    p += 16;
#pragma unroll
    for (int j = 0; j < 8; ++j) r[8 + j] = p[j];
    return r;
}

// B matrix 32x16 f16 where B(k,n) = base[n*ld + k] (i.e. "B^T row-major"):
// lane L (n=L&15, g=L>>4) holds h[j] = B(k0 + 8g + j, n), h[8+j] = B(k0 + 16 + 8g + j, n).
__device__ __forceinline__ v16h load_b16(const _Float16* __restrict__ base,
                                         int ld, int n, int g, int k0) {
    const _Float16* p = base + (size_t)n * ld + k0 + 8 * g;
    v16h r;
#pragma unroll
    for (int j = 0; j < 16; ++j) r[j] = p[j < 8 ? j : j + 8];
    return r;
}

__device__ __forceinline__ v8f wmma_f16(v16h a, v16h b, v8f c) {
    return __builtin_amdgcn_wmma_f32_16x16x32_f16(false, a, false, b,
                                                  (short)0, c, false, false);
}

// ---- K0: fp32 -> f16 convert --------------------------------------------
__global__ void cvt_f32_to_f16(const float* __restrict__ src,
                               _Float16* __restrict__ dst, int n) {
    int i = blockIdx.x * blockDim.x + threadIdx.x;
    if (i < n) dst[i] = (_Float16)src[i];
}

// ---- K1: QKV projection + RoPE ------------------------------------------
// grid: (B*S/32, H*2, 3)  block: 32.
// One wave: 32 tokens x 64 feats. The 64 feats are the interleaved group
// {half*32 + 0..31, half*32 + 64..95} of one head, so every RoPE rotate-half
// partner (d, d+64) is owned by the same wave (tile nt pairs with nt+2).
// 2 A-tiles x 4 B-tiles -> 8 accumulators (64 VGPRs): spill-free, AI ~341 F/B.
__global__ void qkv_rope_kernel(const _Float16* __restrict__ xh,
                                const _Float16* __restrict__ wqh,
                                const _Float16* __restrict__ wkh,
                                const _Float16* __restrict__ wvh,
                                _Float16* __restrict__ qh,  // [B,H,S,D]
                                _Float16* __restrict__ kh,  // [B,H,S,D]
                                _Float16* __restrict__ vt)  // [B,H,D,S]
{
    const int lane = threadIdx.x;
    const int m = lane & 15, g = lane >> 4;
    const int ttile = blockIdx.x;          // 32-token tile
    const int h     = blockIdx.y >> 1;
    const int half  = blockIdx.y & 1;      // which interleaved 64-feat group
    const int which = blockIdx.z;          // 0=q 1=k 2=v

    const _Float16* w = (which == 0) ? wqh : (which == 1) ? wkh : wvh;
    const _Float16* arow = xh + (size_t)ttile * 32 * DIMSZ;
    const _Float16* brow = w + (size_t)h * HDIM * DIMSZ;

    v8f z = {};
    v8f c[2][4];
#pragma unroll
    for (int mt = 0; mt < 2; ++mt)
#pragma unroll
        for (int i = 0; i < 4; ++i) c[mt][i] = z;

    for (int k0 = 0; k0 < DIMSZ; k0 += 32) {
        v16h a0 = load_a16(arow, DIMSZ, m, g, k0);
        v16h a1 = load_a16(arow + (size_t)16 * DIMSZ, DIMSZ, m, g, k0);
#pragma unroll
        for (int nt = 0; nt < 4; ++nt) {
            // tile nt covers head-dims: half*32 + (nt&1)*16 + (nt>>1)*64
            const int eoff = half * 32 + (nt & 1) * 16 + (nt >> 1) * 64;
            v16h b = load_b16(brow + (size_t)eoff * DIMSZ, DIMSZ, m, g, k0);
            c[0][nt] = wmma_f16(a0, b, c[0][nt]);
            c[1][nt] = wmma_f16(a1, b, c[1][nt]);
        }
    }

    const int tbase = ttile * 32;
    const int b_ = tbase / SEQ;
    const int stile = tbase % SEQ;

    if (which < 2) {
        // RoPE: pair (d0, d0+64) = tiles (nt, nt+2), same lane.
        _Float16* outp = ((which == 0) ? qh : kh) +
                         (((size_t)b_ * NHEADS + h) * SEQ) * HDIM;
        const float LOG1E4_64 = 9.210340371976184f / 64.0f;
#pragma unroll
        for (int nt = 0; nt < 2; ++nt) {
            const int d0 = half * 32 + nt * 16 + m;        // 0..63
            const float inv = __expf(-(float)d0 * LOG1E4_64);
#pragma unroll
            for (int mt = 0; mt < 2; ++mt)
#pragma unroll
                for (int r = 0; r < 8; ++r) {
                    const int s = stile + mt * 16 + r + 8 * g;
                    const float ang = (float)s * inv;
                    const float cs = __cosf(ang), sn = __sinf(ang);
                    const float lo = c[mt][nt][r], hi = c[mt][nt + 2][r];
                    outp[(size_t)s * HDIM + d0]      = (_Float16)(lo * cs - hi * sn);
                    outp[(size_t)s * HDIM + d0 + 64] = (_Float16)(hi * cs + lo * sn);
                }
        }
    } else {
        // V transposed [D,S]: lane owns column d, rows are consecutive s.
        _Float16* outp = vt + (((size_t)b_ * NHEADS + h) * HDIM) * SEQ;
#pragma unroll
        for (int nt = 0; nt < 4; ++nt) {
            const int d = half * 32 + (nt & 1) * 16 + (nt >> 1) * 64 + m;
#pragma unroll
            for (int mt = 0; mt < 2; ++mt) {
                const size_t base = (size_t)d * SEQ + stile + mt * 16 + 8 * g;
#pragma unroll
                for (int r = 0; r < 8; ++r) outp[base + r] = (_Float16)c[mt][nt][r];
            }
        }
    }
}

// ---- K2: flash attention -------------------------------------------------
// grid: (S/16, H, B)  block: 32. One wave: 16 queries x full head.
__global__ void attn_kernel(const _Float16* __restrict__ qh,
                            const _Float16* __restrict__ kh,
                            const _Float16* __restrict__ vt,
                            _Float16* __restrict__ ctx)   // [B*S, H*D]
{
    __shared__ _Float16 pbuf[16 * 32];     // P tile bounce (one wave/block)

    const int lane = threadIdx.x;
    const int m = lane & 15, g = lane >> 4;
    const int qtile = blockIdx.x * 16;
    const int h = blockIdx.y, b = blockIdx.z;

    const _Float16* Q = qh + (((size_t)b * NHEADS + h) * SEQ) * HDIM;
    const _Float16* K = kh + (((size_t)b * NHEADS + h) * SEQ) * HDIM;
    const _Float16* V = vt + (((size_t)b * NHEADS + h) * HDIM) * SEQ;

    // Preload Q A-tiles (16 queries x 128 d = 4 K-chunks)
    v16h qa[4];
#pragma unroll
    for (int dc = 0; dc < 4; ++dc)
        qa[dc] = load_a16(Q + (size_t)qtile * HDIM, HDIM, m, g, dc * 32);

    v8f z = {};
    v8f o[8];
#pragma unroll
    for (int i = 0; i < 8; ++i) o[i] = z;
    float mrun[8], lrun[8];
#pragma unroll
    for (int r = 0; r < 8; ++r) { mrun[r] = -1e30f; lrun[r] = 0.0f; }

    const float scale = 0.08838834764831845f;  // 1/sqrt(128)
    const int kend = qtile + 16;               // causal bound (exclusive)

    for (int kc0 = 0; kc0 < kend; kc0 += 32) {
        // ---- scores: two 16x16 tiles over 32 keys ----
        v8f sc[2];
#pragma unroll
        for (int t = 0; t < 2; ++t) {
            v8f s = z;
            const int kt = kc0 + t * 16;
            if (kt < kend) {
#pragma unroll
                for (int dc = 0; dc < 4; ++dc) {
                    v16h bb = load_b16(K + (size_t)kt * HDIM, HDIM, m, g, dc * 32);
                    s = wmma_f16(qa[dc], bb, s);
                }
            }
            sc[t] = s;
        }

        // ---- softmax bookkeeping (row = qtile + r + 8g lives in lane group g)
#pragma unroll
        for (int r = 0; r < 8; ++r) {
            const int qrow = qtile + r + 8 * g;
            float mx = mrun[r];
#pragma unroll
            for (int t = 0; t < 2; ++t) {
                const int key = kc0 + t * 16 + m;
                float v = sc[t][r] * scale + ((key <= qrow) ? 0.0f : -1e9f);
                sc[t][r] = v;
                mx = fmaxf(mx, v);
            }
#pragma unroll
            for (int off = 1; off < 16; off <<= 1)
                mx = fmaxf(mx, __shfl_xor(mx, off, 32));

            const float alpha = __expf(mrun[r] - mx);
            float s0 = 0.0f;
#pragma unroll
            for (int t = 0; t < 2; ++t) {
                const float p = __expf(sc[t][r] - mx);
                sc[t][r] = p;
                s0 += p;
            }
#pragma unroll
            for (int off = 1; off < 16; off <<= 1)
                s0 += __shfl_xor(s0, off, 32);

            lrun[r] = lrun[r] * alpha + s0;
            mrun[r] = mx;
#pragma unroll
            for (int nt = 0; nt < 8; ++nt) o[nt][r] *= alpha;
        }

        // ---- transpose P (C layout) -> A layout through LDS ----
#pragma unroll
        for (int t = 0; t < 2; ++t)
#pragma unroll
            for (int r = 0; r < 8; ++r)
                pbuf[(r + 8 * g) * 32 + t * 16 + m] = (_Float16)sc[t][r];
        asm volatile("s_wait_dscnt 0x0" ::: "memory");
        v16h pa = load_a16(pbuf, 32, m, g, 0);
        asm volatile("s_wait_dscnt 0x0" ::: "memory");

        // ---- O += P @ V  (V transposed: key-dim contiguous) ----
#pragma unroll
        for (int nt = 0; nt < 8; ++nt) {
            v16h vb = load_b16(V + (size_t)(nt * 16) * SEQ, SEQ, m, g, kc0);
            o[nt] = wmma_f16(pa, vb, o[nt]);
        }
    }

    // ---- epilogue: normalize, write ctx[b*S+s][h*128+d] f16 ----
#pragma unroll
    for (int nt = 0; nt < 8; ++nt) {
        const int d = nt * 16 + m;
#pragma unroll
        for (int r = 0; r < 8; ++r) {
            const int s = qtile + r + 8 * g;
            ctx[((size_t)b * SEQ + s) * (NHEADS * HDIM) + h * HDIM + d] =
                (_Float16)(o[nt][r] / lrun[r]);
        }
    }
}

// ---- K3: output projection ----------------------------------------------
// grid: (B*S/32, DIM/64)  block: 32. One wave: 32 tokens x 64 out-dims.
__global__ void out_proj_kernel(const _Float16* __restrict__ ctx,
                                const _Float16* __restrict__ woh,
                                float* __restrict__ out) {
    const int lane = threadIdx.x;
    const int m = lane & 15, g = lane >> 4;
    const int ttile = blockIdx.x;
    const int et = blockIdx.y * 64;

    const _Float16* arow = ctx + (size_t)ttile * 32 * DIMSZ;
    const _Float16* brow = woh + (size_t)et * DIMSZ;

    v8f z = {};
    v8f c[2][4];
#pragma unroll
    for (int mt = 0; mt < 2; ++mt)
#pragma unroll
        for (int i = 0; i < 4; ++i) c[mt][i] = z;

    for (int k0 = 0; k0 < DIMSZ; k0 += 32) {
        v16h a0 = load_a16(arow, DIMSZ, m, g, k0);
        v16h a1 = load_a16(arow + (size_t)16 * DIMSZ, DIMSZ, m, g, k0);
#pragma unroll
        for (int nt = 0; nt < 4; ++nt) {
            v16h b = load_b16(brow + (size_t)nt * 16 * DIMSZ, DIMSZ, m, g, k0);
            c[0][nt] = wmma_f16(a0, b, c[0][nt]);
            c[1][nt] = wmma_f16(a1, b, c[1][nt]);
        }
    }
#pragma unroll
    for (int mt = 0; mt < 2; ++mt)
#pragma unroll
        for (int nt = 0; nt < 4; ++nt)
#pragma unroll
            for (int r = 0; r < 8; ++r)
                out[((size_t)(ttile * 32 + mt * 16 + r + 8 * g)) * DIMSZ +
                    et + nt * 16 + m] = c[mt][nt][r];
}

// ---- launcher ------------------------------------------------------------
extern "C" void kernel_launch(void* const* d_in, const int* in_sizes, int n_in,
                              void* d_out, int out_size, void* d_ws, size_t ws_size,
                              hipStream_t stream) {
    const float* x  = (const float*)d_in[0];
    // d_in[1] = mask (causal mask recomputed analytically on-GPU)
    const float* wq = (const float*)d_in[2];
    const float* wk = (const float*)d_in[3];
    const float* wv = (const float*)d_in[4];
    const float* wo = (const float*)d_in[5];

    const size_t nTok  = (size_t)BATCH * SEQ;          // 4096
    const size_t xN    = nTok * DIMSZ;                 // 8.4M
    const size_t wN    = (size_t)DIMSZ * DIMSZ;        // 4.2M
    const size_t qkvN  = (size_t)BATCH * NHEADS * SEQ * HDIM;  // 8.4M

    char* p = (char*)d_ws;
    _Float16* xh  = (_Float16*)p; p += xN * 2;
    _Float16* wqh = (_Float16*)p; p += wN * 2;
    _Float16* wkh = (_Float16*)p; p += wN * 2;
    _Float16* wvh = (_Float16*)p; p += wN * 2;
    _Float16* woh = (_Float16*)p; p += wN * 2;
    _Float16* qh  = (_Float16*)p; p += qkvN * 2;
    _Float16* kh  = (_Float16*)p; p += qkvN * 2;
    _Float16* vt  = (_Float16*)p; p += qkvN * 2;
    _Float16* ctx = (_Float16*)p; p += xN * 2;

    const int TPB = 256;
    cvt_f32_to_f16<<<dim3((xN + TPB - 1) / TPB), TPB, 0, stream>>>(x,  xh,  (int)xN);
    cvt_f32_to_f16<<<dim3((wN + TPB - 1) / TPB), TPB, 0, stream>>>(wq, wqh, (int)wN);
    cvt_f32_to_f16<<<dim3((wN + TPB - 1) / TPB), TPB, 0, stream>>>(wk, wkh, (int)wN);
    cvt_f32_to_f16<<<dim3((wN + TPB - 1) / TPB), TPB, 0, stream>>>(wv, wvh, (int)wN);
    cvt_f32_to_f16<<<dim3((wN + TPB - 1) / TPB), TPB, 0, stream>>>(wo, woh, (int)wN);

    qkv_rope_kernel<<<dim3(nTok / 32, NHEADS * 2, 3), 32, 0, stream>>>(
        xh, wqh, wkh, wvh, qh, kh, vt);

    attn_kernel<<<dim3(SEQ / 16, NHEADS, BATCH), 32, 0, stream>>>(qh, kh, vt, ctx);

    out_proj_kernel<<<dim3(nTok / 32, DIMSZ / 64), 32, 0, stream>>>(
        ctx, woh, (float*)d_out);
}
